// SleepGATNet_43602507989152
// MI455X (gfx1250) — matrix operand order repaired and checked by reading, not compiled, not run on hardware
//
#include <hip/hip_runtime.h>

// ===========================================================================
// SleepGATNet for MI455X (gfx1250): f16 WMMA implicit-GEMM conv stack
// (>=95% of FLOPs) + scalar f32 heads.  All activations f16 position-major.
// ===========================================================================

typedef _Float16 h16;
typedef __attribute__((ext_vector_type(16))) _Float16 hv16;
typedef __attribute__((ext_vector_type(8)))  _Float16 hv8;
typedef __attribute__((ext_vector_type(8)))  float    fv8;

#define RSBN 0.9999950000374997f   // 1/sqrt(1+1e-5)

// --------------------------------------------------------------------------
// cast x (f32) -> f16
// --------------------------------------------------------------------------
__global__ void cast_x_kernel(const float* __restrict__ x, h16* __restrict__ xh, long n) {
  long i = (long)blockIdx.x * blockDim.x + threadIdx.x;
  if (i < n) xh[i] = (h16)x[i];
}

// --------------------------------------------------------------------------
// Pack conv weights (f32, [Cout][Cin][k]) into WMMA A-fragment order (f16),
// folding the BatchNorm scale g*RSBN into the weights.
// Fragment f = ((ot*k + t)*(Cin/32) + icc); 512 halves per fragment:
//   lane l (0..31), slot j (0..15):
//     M = l&15 ; K = ((l>>4)<<4) + j
// --------------------------------------------------------------------------
__global__ void pack_w_kernel(const float* __restrict__ w, const float* __restrict__ g,
                              h16* __restrict__ dst, int Cout, int Cin, int k) {
  int tid = blockIdx.x * 256 + threadIdx.x;
  int total = Cout * Cin * k;
  if (tid >= total) return;
  int nIcc = Cin >> 5;
  int f    = tid >> 9;
  int rem  = tid & 511;
  int l    = rem >> 4, j = rem & 15;
  int icc  = f % nIcc;
  int tmp  = f / nIcc;
  int t    = tmp % k;
  int ot   = tmp / k;
  int K    = ((l >> 4) << 4) + j;
  int oc   = (ot << 4) + (l & 15);
  int ic   = (icc << 5) + K;
  float s  = g[oc] * RSBN;
  dst[tid] = (h16)(w[((size_t)oc * Cin + ic) * k + t] * s);
}

// --------------------------------------------------------------------------
// Layer-1 conv1 (Cin=1, k=50, stride=5, pad=25) + BN + ReLU  -> y1 f16
// y1 layout [sig][601][64]
// --------------------------------------------------------------------------
__global__ void l1_conv1_kernel(const float* __restrict__ x,
                                const float* __restrict__ w1, const float* __restrict__ b1,
                                const float* __restrict__ g1, const float* __restrict__ be1,
                                h16* __restrict__ y1) {
  int sig = blockIdx.x;
  int pos = blockIdx.y * 4 + (threadIdx.x >> 6);
  int oc  = threadIdx.x & 63;
  if (pos >= 601) return;
  const float* xs = x + (size_t)sig * 3000;
  const float* wr = w1 + oc * 50;
  int base = pos * 5 - 25;
  float acc = 0.f;
  for (int t = 0; t < 50; ++t) {
    int ix = base + t;
    float xv = (ix >= 0 && ix < 3000) ? xs[ix] : 0.f;
    acc += wr[t] * xv;
  }
  float s = g1[oc] * RSBN;
  float v = s * (acc + b1[oc]) + be1[oc];
  v = v > 0.f ? v : 0.f;
  y1[((size_t)sig * 601 + pos) * 64 + oc] = (h16)v;
}

// --------------------------------------------------------------------------
// Generic WMMA conv1d (+BN, optional residual shortcut conv, ReLU).
// One wave per 16(oc) x 16(pos) tile.  Input f16 position-major.
// scMode: 0 = none (mid conv), 1 = scalar 1-channel shortcut (layer1),
//         2 = WMMA shortcut (Csc multiple of 32)
// --------------------------------------------------------------------------
struct ConvArgs {
  const h16* in;  h16* out;  const h16* wpk;
  const float* bc; const float* gc; const float* bec;
  int Cin, Lin, Cout, Lout, k, stride, dil, pad;
  int scMode;
  const h16* scIn; const h16* scWpk; const float* scWraw;
  const float* bs2; const float* gs2; const float* bes2;
  int Csc, Lsc, scStride;
};

__global__ __launch_bounds__(32) void wmma_conv_kernel(ConvArgs A) {
  __shared__ h16 lds[5120];                 // max R*Cin = 39*128 = 4992 halves
  const int sig  = blockIdx.x;
  const int nOc  = A.Cout >> 4;
  const int ot   = (int)blockIdx.y % nOc;
  const int pt   = (int)blockIdx.y / nOc;
  const int p0   = pt << 4;
  const int lane = threadIdx.x;
  const int lo   = lane & 15, hi = lane >> 4;

  // ---- stage input window [R x Cin] into LDS (zero-fill out of range) ----
  const int R      = 15 * A.stride + (A.k - 1) * A.dil + 1;
  const int uPer   = A.Cin >> 3;             // 16B units per row
  const int pstart = p0 * A.stride - A.pad;
  const int total  = R * uPer;
  for (int idx = lane; idx < total; idx += 32) {
    int c = idx / uPer;
    int u = idx - c * uPer;
    int prow = pstart + c;
    uint4 val = make_uint4(0u, 0u, 0u, 0u);
    if (prow >= 0 && prow < A.Lin)
      val = *(const uint4*)(A.in + ((size_t)sig * A.Lin + prow) * A.Cin + (u << 3));
    *(uint4*)(&lds[c * A.Cin + (u << 3)]) = val;
  }
  asm volatile("s_wait_dscnt 0" ::: "memory");

  // ---- main conv: K loop over taps x 32-channel chunks ----
  fv8 acc = {};
  const int nIcc = A.Cin >> 5;
  for (int t = 0; t < A.k; ++t) {
    const int col     = lo * A.stride + t * A.dil;
    const h16* brow   = &lds[col * A.Cin + (hi << 4)];
    const h16* afrag  = A.wpk + ((size_t)(ot * A.k + t) * nIcc) * 512 + (lane << 4);
    if (t + 1 < A.k)
      __builtin_prefetch(A.wpk + ((size_t)(ot * A.k + t + 1) * nIcc) * 512 + (lane << 4), 0, 1);
    for (int icc = 0; icc < nIcc; ++icc) {
      hv16 b = *(const hv16*)(brow + (icc << 5));          // K = hi*16 + j contiguous ic
      hv16 a = *(const hv16*)(afrag + (size_t)icc * 512);
      acc = __builtin_amdgcn_wmma_f32_16x16x32_f16(false, a, false, b, (short)0, acc,
                                                   false, false);
    }
  }

  // ---- residual shortcut ----
  fv8 acc2 = {};
  float scScalar = 0.f;
  if (A.scMode == 2) {
    const bool ok  = (p0 + lo) < A.Lout;
    const int spos = (p0 + lo) * A.scStride;
    const int nS   = A.Csc >> 5;
    for (int icc = 0; icc < nS; ++icc) {
      hv16 b = {};
      if (ok)
        b = *(const hv16*)(A.scIn + ((size_t)sig * A.Lsc + spos) * A.Csc + (icc << 5) + (hi << 4));
      hv16 a = *(const hv16*)(A.scWpk + ((size_t)(ot * nS + icc)) * 512 + (lane << 4));
      acc2 = __builtin_amdgcn_wmma_f32_16x16x32_f16(false, a, false, b, (short)0, acc2,
                                                    false, false);
    }
  } else if (A.scMode == 1) {
    const int pos = p0 + lo;
    scScalar = (pos < A.Lout) ? (float)A.scIn[(size_t)sig * A.Lsc + (size_t)pos * A.scStride] : 0.f;
  }

  // ---- epilogue: BN biases, residual add, ReLU, vector f16 store ----
  const int pos = p0 + lo;
  if (pos < A.Lout) {
    hv8 ov = {};
    for (int v = 0; v < 8; ++v) {
      const int oc = (ot << 4) + (hi << 3) + v;
      const float s1 = A.gc[oc] * RSBN;
      float val = acc[v] + s1 * A.bc[oc] + A.bec[oc];
      if (A.scMode == 2) {
        const float s2 = A.gs2[oc] * RSBN;
        val += acc2[v] + s2 * A.bs2[oc] + A.bes2[oc];
      } else if (A.scMode == 1) {
        const float s2 = A.gs2[oc] * RSBN;
        val += s2 * (A.scWraw[oc] * scScalar + A.bs2[oc]) + A.bes2[oc];
      }
      val = val > 0.f ? val : 0.f;
      ov[v] = (h16)val;
    }
    *(hv8*)(A.out + ((size_t)sig * A.Lout + pos) * A.Cout + (ot << 4) + (hi << 3)) = ov;
  }
}

// --------------------------------------------------------------------------
// AdaptiveAvgPool(150) + FC(256->128) + ReLU  -> feat f32 [1920][128]
// --------------------------------------------------------------------------
__global__ void pool_fc_kernel(const h16* __restrict__ st3, const float* __restrict__ fcw,
                               const float* __restrict__ fcb, float* __restrict__ feat) {
  int sig = blockIdx.x, t = threadIdx.x;      // 256 threads
  __shared__ float m[256];
  const h16* base = st3 + (size_t)sig * 150 * 256;
  float s = 0.f;
  for (int p = 0; p < 150; ++p) s += (float)base[p * 256 + t];
  m[t] = s * (1.f / 150.f);
  __syncthreads();
  if (t < 128) {
    const float* wr = fcw + t * 256;
    float a = fcb[t];
    for (int kk = 0; kk < 256; ++kk) a += m[kk] * wr[kk];
    feat[(size_t)sig * 128 + t] = a > 0.f ? a : 0.f;
  }
}

// --------------------------------------------------------------------------
// Small setup: prior matrix log(A_prior+1e-9) (36) and TAGAT time-enc dot
// tw[i][j] = (te[i][j] . wt) (36).  One block, 64 threads.
// --------------------------------------------------------------------------
__global__ void setup_kernel(const float* __restrict__ a_plv, const float* __restrict__ wf,
                             const float* __restrict__ E,
                             const float* __restrict__ pos, const float* __restrict__ t1w,
                             const float* __restrict__ t1b, const float* __restrict__ t2w,
                             const float* __restrict__ t2b, const float* __restrict__ attT,
                             float* __restrict__ logAp, float* __restrict__ twout) {
  __shared__ float Asp[36];
  int t = threadIdx.x;
  float e0 = expf(wf[0]), e1 = expf(wf[1]);
  float w0 = e0 / (e0 + e1), w1 = e1 / (e0 + e1);
  const float CO[6][3] = {{-0.707f, 0.f, 0.707f}, {0.707f, 0.f, 0.707f},
                          {-0.4f, -0.9f, 0.1f},   {0.4f, -0.9f, 0.1f},
                          {-0.4f, 0.8f, 0.4f},    {0.4f, 0.8f, 0.4f}};
  if (t < 36) {
    float sm = 0.f;
    for (int b = 0; b < 320; ++b) sm += a_plv[b * 36 + t];
    sm *= (1.f / 320.f);
    int i = t / 6, j = t % 6;
    float d2 = 0.f;
    for (int q = 0; q < 3; ++q) { float d = CO[i][q] - CO[j][q]; d2 += d * d; }
    float a0 = w0 * expf(-d2) + w1 * sm;
    float e  = E[t];
    float sp = (e > 20.f) ? e : log1pf(expf(e));
    Asp[t] = a0 * sp;
  }
  __syncthreads();
  if (t < 36) {
    int i = t / 6, j = t % 6;
    float a = 0.5f * (Asp[t] + Asp[j * 6 + i]);
    if (i == j) a = 0.f;
    float ap = 1.f / (1.f + expf(-10.f * (a - 0.5f)));
    logAp[t] = logf(ap + 1e-9f);
    // time encoding term
    float tau = pos[i] - pos[j];
    float u[32];
    for (int kk = 0; kk < 32; ++kk) { float z = tau * t1w[kk] + t1b[kk]; u[kk] = z > 0.f ? z : 0.f; }
    float accv = 0.f;
    for (int mq = 0; mq < 32; ++mq) {
      float z = t2b[mq];
      for (int kk = 0; kk < 32; ++kk) z += u[kk] * t2w[mq * 32 + kk];
      accv += z * attT[128 + mq];
    }
    twout[t] = accv;
  }
}

// --------------------------------------------------------------------------
// TAGAT + SCGAT + LayerNorms + GatedFusion (per b*s block, 256 threads)
// writes s_logits to d_out[80 + bs*5], fused[bs][256] to ws.
// --------------------------------------------------------------------------
__global__ void gat_fuse_kernel(const float* __restrict__ feat,
    const float* __restrict__ twp, const float* __restrict__ tbp, const float* __restrict__ tatt,
    const float* __restrict__ swp, const float* __restrict__ sbp, const float* __restrict__ satt,
    const float* __restrict__ lam,
    const float* __restrict__ sm1w, const float* __restrict__ sm1b,
    const float* __restrict__ sm2w, const float* __restrict__ sm2b,
    const float* __restrict__ mc1w, const float* __restrict__ mc1b,
    const float* __restrict__ mc2w, const float* __restrict__ mc2b,
    const float* __restrict__ ntg, const float* __restrict__ ntb,
    const float* __restrict__ nsg, const float* __restrict__ nsb,
    const float* __restrict__ gw, const float* __restrict__ gb,
    const float* __restrict__ logAp, const float* __restrict__ tw,
    float* __restrict__ fused, float* __restrict__ outSlog) {
  int bs = blockIdx.x, t = threadIdx.x;   // 256 threads
  __shared__ float xp[1536], xq[1536], to_[1536], so_[1536];
  __shared__ float al[144];
  __shared__ float fmean[128], h1[128], slog[8], sprob[8], h2[64], cvec[64];
  __shared__ float ssA[24], sdA[24], ssB[24], sdB[24];
  __shared__ float red[256], x1v[256], x2v[256], scal[4];
  const float* fb = feat + (size_t)bs * 6 * 128;

  // projections xp (TAGAT) and xq (SCGAT): [c][256]
  {
    const float* wr1 = twp + t * 128;
    const float* wr2 = swp + t * 128;
    for (int c = 0; c < 6; ++c) {
      const float* fr = fb + c * 128;
      float a1 = tbp[t], a2 = sbp[t];
      for (int kk = 0; kk < 128; ++kk) { float fv = fr[kk]; a1 += fv * wr1[kk]; a2 += fv * wr2[kk]; }
      xp[c * 256 + t] = a1; xq[c * 256 + t] = a2;
    }
  }
  if (t < 128) {
    float s = 0.f;
    for (int c = 0; c < 6; ++c) s += fb[c * 128 + t];
    fmean[t] = s * (1.f / 6.f);
  }
  __syncthreads();
  if (t < 128) {
    const float* wr = sm1w + t * 128;
    float a = sm1b[t];
    for (int kk = 0; kk < 128; ++kk) a += fmean[kk] * wr[kk];
    h1[t] = a > 0.f ? a : 0.f;
  }
  __syncthreads();
  if (t < 5) {
    const float* wr = sm2w + t * 128;
    float a = sm2b[t];
    for (int kk = 0; kk < 128; ++kk) a += h1[kk] * wr[kk];
    slog[t] = a;
    outSlog[bs * 5 + t] = a;          // s_logits output
  }
  __syncthreads();
  if (t == 0) {                       // softmax(s_logits)
    float mx = slog[0];
    for (int q = 1; q < 5; ++q) mx = fmaxf(mx, slog[q]);
    float sm = 0.f, ev[5];
    for (int q = 0; q < 5; ++q) { ev[q] = expf(slog[q] - mx); sm += ev[q]; }
    for (int q = 0; q < 5; ++q) sprob[q] = ev[q] / sm;
  }
  __syncthreads();
  if (t < 64) {
    const float* wr = mc1w + t * 5;
    float a = mc1b[t];
    for (int kk = 0; kk < 5; ++kk) a += sprob[kk] * wr[kk];
    h2[t] = a > 0.f ? a : 0.f;
  }
  __syncthreads();
  if (t < 64) {
    const float* wr = mc2w + t * 64;
    float a = mc2b[t];
    for (int kk = 0; kk < 64; ++kk) a += h2[kk] * wr[kk];
    cvec[t] = a;
  }
  __syncthreads();
  if (t == 0) {
    float a = 0.f;
    for (int kk = 0; kk < 64; ++kk) a += cvec[kk] * satt[128 + kk];
    scal[0] = a;                      // c . wc2
  }
  if (t < 24) {                       // per (node j, head h) src/dst scores
    int j = t >> 2, h = t & 3;
    const float* v1 = xp + j * 256 + h * 64;
    const float* v2 = xq + j * 256 + h * 64;
    float s1 = 0, d1 = 0, s2 = 0, d2 = 0;
    for (int kk = 0; kk < 64; ++kk) {
      s1 += v1[kk] * tatt[kk];      d1 += v1[kk] * tatt[64 + kk];
      s2 += v2[kk] * satt[kk];      d2 += v2[kk] * satt[64 + kk];
    }
    ssA[t] = s1; sdA[t] = d1; ssB[t] = s2; sdB[t] = d2;
  }
  __syncthreads();

  // ---- TAGAT attention ----
  if (t < 24) {
    int i = t >> 2, h = t & 3;
    float ev[6], mx = -1e30f;
    for (int j = 0; j < 6; ++j) {
      float e = ssA[j * 4 + h] + sdA[i * 4 + h] + tw[i * 6 + j];
      e = e > 0.f ? e : 0.2f * e;
      ev[j] = e; mx = fmaxf(mx, e);
    }
    float sm = 0.f;
    for (int j = 0; j < 6; ++j) { ev[j] = expf(ev[j] - mx); sm += ev[j]; }
    float inv = 1.f / sm;
    for (int j = 0; j < 6; ++j) al[(i * 4 + h) * 6 + j] = ev[j] * inv;
  }
  __syncthreads();
  {
    int h = t >> 6, d = t & 63;
    for (int i = 0; i < 6; ++i) {
      float a = 0.f;
      for (int j = 0; j < 6; ++j) a += al[(i * 4 + h) * 6 + j] * xp[j * 256 + h * 64 + d];
      to_[i * 256 + h * 64 + d] = a;
    }
  }
  __syncthreads();

  auto bred = [&](float v) -> float {
    red[t] = v; __syncthreads();
    for (int s = 128; s > 0; s >>= 1) { if (t < s) red[t] += red[t + s]; __syncthreads(); }
    float r = red[0]; __syncthreads(); return r;
  };
  // LayerNorm (norm_t) per node
  for (int i = 0; i < 6; ++i) {
    float v = to_[i * 256 + t];
    float mean = bred(v) * (1.f / 256.f);
    float dv = v - mean;
    float var = bred(dv * dv) * (1.f / 256.f);
    to_[i * 256 + t] = dv * rsqrtf(var + 1e-5f) * ntg[t] + ntb[t];
  }
  __syncthreads();
  {
    float s = 0.f;
    for (int i = 0; i < 6; ++i) s += to_[i * 256 + t];
    x1v[t] = s * (1.f / 6.f);
  }
  __syncthreads();

  // ---- SCGAT attention ----
  if (t < 24) {
    int i = t >> 2, h = t & 3;
    float ev[6], mx = -1e30f;
    for (int j = 0; j < 6; ++j) {
      float e = ssB[j * 4 + h] + sdB[i * 4 + h] + scal[0];
      e = e > 0.f ? e : 0.2f * e;                 // leaky first
      e += lam[h] * logAp[i * 6 + j];             // then prior bias
      ev[j] = e; mx = fmaxf(mx, e);
    }
    float sm = 0.f;
    for (int j = 0; j < 6; ++j) { ev[j] = expf(ev[j] - mx); sm += ev[j]; }
    float inv = 1.f / sm;
    for (int j = 0; j < 6; ++j) al[(i * 4 + h) * 6 + j] = ev[j] * inv;
  }
  __syncthreads();
  {
    int h = t >> 6, d = t & 63;
    for (int i = 0; i < 6; ++i) {
      float a = 0.f;
      for (int j = 0; j < 6; ++j) a += al[(i * 4 + h) * 6 + j] * xq[j * 256 + h * 64 + d];
      so_[i * 256 + h * 64 + d] = a;
    }
  }
  __syncthreads();
  for (int i = 0; i < 6; ++i) {
    float v = so_[i * 256 + t];
    float mean = bred(v) * (1.f / 256.f);
    float dv = v - mean;
    float var = bred(dv * dv) * (1.f / 256.f);
    so_[i * 256 + t] = dv * rsqrtf(var + 1e-5f) * nsg[t] + nsb[t];
  }
  __syncthreads();
  {
    float s = 0.f;
    for (int i = 0; i < 6; ++i) s += so_[i * 256 + t];
    x2v[t] = s * (1.f / 6.f);
  }
  __syncthreads();

  // ---- gated fusion ----
  {
    const float* wr = gw + t * 512;
    float a = gb[t];
    for (int kk = 0; kk < 256; ++kk) a += x1v[kk] * wr[kk];
    for (int kk = 0; kk < 256; ++kk) a += x2v[kk] * wr[256 + kk];
    float z = 1.f / (1.f + expf(-a));
    fused[(size_t)bs * 256 + t] = z * x1v[t] + (1.f - z) * x2v[t];
  }
}

// --------------------------------------------------------------------------
// BiLSTM over 20 epochs + classifier at center epoch (per batch block)
// --------------------------------------------------------------------------
__global__ void lstm_cls_kernel(const float* __restrict__ fused,
    const float* __restrict__ wihf, const float* __restrict__ whhf,
    const float* __restrict__ bihf, const float* __restrict__ bhhf,
    const float* __restrict__ wihb, const float* __restrict__ whhb,
    const float* __restrict__ bihb, const float* __restrict__ bhhb,
    const float* __restrict__ cw, const float* __restrict__ cb,
    float* __restrict__ outLogits) {
  int b = blockIdx.x, t = threadIdx.x;   // 128 threads
  __shared__ float h[128], cs[128], xv[256], hf10[128], hb10[128];
  h[t] = 0.f; cs[t] = 0.f; __syncthreads();
  for (int s = 0; s <= 10; ++s) {        // forward: only need state at center
    const float* xr = fused + ((size_t)b * 20 + s) * 256;
    xv[t] = xr[t]; xv[t + 128] = xr[t + 128];
    __syncthreads();
    float g4[4];
    for (int q = 0; q < 4; ++q) {
      int row = q * 128 + t;
      const float* wi = wihf + (size_t)row * 256;
      const float* wh = whhf + (size_t)row * 128;
      float a = bihf[row] + bhhf[row];
      for (int kk = 0; kk < 256; ++kk) a += xv[kk] * wi[kk];
      for (int kk = 0; kk < 128; ++kk) a += h[kk] * wh[kk];
      g4[q] = a;
    }
    __syncthreads();
    float ig = 1.f / (1.f + expf(-g4[0]));
    float fg = 1.f / (1.f + expf(-g4[1]));
    float gg = tanhf(g4[2]);
    float og = 1.f / (1.f + expf(-g4[3]));
    float cn = fg * cs[t] + ig * gg;
    cs[t] = cn; h[t] = og * tanhf(cn);
    __syncthreads();
  }
  hf10[t] = h[t];
  __syncthreads();
  h[t] = 0.f; cs[t] = 0.f; __syncthreads();
  for (int s = 19; s >= 10; --s) {       // backward: 19 down to center
    const float* xr = fused + ((size_t)b * 20 + s) * 256;
    xv[t] = xr[t]; xv[t + 128] = xr[t + 128];
    __syncthreads();
    float g4[4];
    for (int q = 0; q < 4; ++q) {
      int row = q * 128 + t;
      const float* wi = wihb + (size_t)row * 256;
      const float* wh = whhb + (size_t)row * 128;
      float a = bihb[row] + bhhb[row];
      for (int kk = 0; kk < 256; ++kk) a += xv[kk] * wi[kk];
      for (int kk = 0; kk < 128; ++kk) a += h[kk] * wh[kk];
      g4[q] = a;
    }
    __syncthreads();
    float ig = 1.f / (1.f + expf(-g4[0]));
    float fg = 1.f / (1.f + expf(-g4[1]));
    float gg = tanhf(g4[2]);
    float og = 1.f / (1.f + expf(-g4[3]));
    float cn = fg * cs[t] + ig * gg;
    cs[t] = cn; h[t] = og * tanhf(cn);
    __syncthreads();
  }
  hb10[t] = h[t];
  __syncthreads();
  if (t < 5) {
    const float* wr = cw + t * 256;
    float a = cb[t];
    for (int kk = 0; kk < 128; ++kk) a += hf10[kk] * wr[kk] + hb10[kk] * wr[128 + kk];
    outLogits[b * 5 + t] = a;
  }
}

// ==========================================================================
// Host launcher
// ==========================================================================
extern "C" void kernel_launch(void* const* d_in, const int* in_sizes, int n_in,
                              void* d_out, int out_size, void* d_ws, size_t ws_size,
                              hipStream_t stream) {
  if (n_in < 78) return;
  // ---- input mapping (setup_inputs dict insertion order, recursive) ----
  const float* x     = (const float*)d_in[0];
  const float* a_plv = (const float*)d_in[1];
  const float *l1w1 = (const float*)d_in[2],  *l1b1 = (const float*)d_in[3],
              *l1g1 = (const float*)d_in[4],  *l1be1 = (const float*)d_in[5],
              *l1w2 = (const float*)d_in[6],  *l1b2 = (const float*)d_in[7],
              *l1g2 = (const float*)d_in[8],  *l1be2 = (const float*)d_in[9],
              *l1ws = (const float*)d_in[10], *l1bs = (const float*)d_in[11],
              *l1gs = (const float*)d_in[12], *l1bes = (const float*)d_in[13];
  const float *l2w1 = (const float*)d_in[14], *l2b1 = (const float*)d_in[15],
              *l2g1 = (const float*)d_in[16], *l2be1 = (const float*)d_in[17],
              *l2w2 = (const float*)d_in[18], *l2b2 = (const float*)d_in[19],
              *l2g2 = (const float*)d_in[20], *l2be2 = (const float*)d_in[21],
              *l2ws = (const float*)d_in[22], *l2bs = (const float*)d_in[23],
              *l2gs = (const float*)d_in[24], *l2bes = (const float*)d_in[25];
  const float *l3w1 = (const float*)d_in[26], *l3b1 = (const float*)d_in[27],
              *l3g1 = (const float*)d_in[28], *l3be1 = (const float*)d_in[29],
              *l3w2 = (const float*)d_in[30], *l3b2 = (const float*)d_in[31],
              *l3g2 = (const float*)d_in[32], *l3be2 = (const float*)d_in[33],
              *l3ws = (const float*)d_in[34], *l3bs = (const float*)d_in[35],
              *l3gs = (const float*)d_in[36], *l3bes = (const float*)d_in[37];
  const float *fcw = (const float*)d_in[38], *fcb = (const float*)d_in[39];
  const float *pwf = (const float*)d_in[40], *pE = (const float*)d_in[41];
  const float *twp = (const float*)d_in[42], *tbp = (const float*)d_in[43],
              *tpos = (const float*)d_in[44], *t1w = (const float*)d_in[45],
              *t1b = (const float*)d_in[46],  *t2w = (const float*)d_in[47],
              *t2b = (const float*)d_in[48],  *tatt = (const float*)d_in[49];
  const float *swp = (const float*)d_in[50], *sbp = (const float*)d_in[51],
              *sm1w = (const float*)d_in[52], *sm1b = (const float*)d_in[53],
              *sm2w = (const float*)d_in[54], *sm2b = (const float*)d_in[55],
              *mc1w = (const float*)d_in[56], *mc1b = (const float*)d_in[57],
              *mc2w = (const float*)d_in[58], *mc2b = (const float*)d_in[59],
              *satt = (const float*)d_in[60], *lam = (const float*)d_in[61];
  const float *ntg = (const float*)d_in[62], *ntb = (const float*)d_in[63];
  const float *nsg = (const float*)d_in[64], *nsb = (const float*)d_in[65];
  const float *gw = (const float*)d_in[66], *gb = (const float*)d_in[67];
  const float *wihf = (const float*)d_in[68], *whhf = (const float*)d_in[69],
              *bihf = (const float*)d_in[70], *bhhf = (const float*)d_in[71],
              *wihb = (const float*)d_in[72], *whhb = (const float*)d_in[73],
              *bihb = (const float*)d_in[74], *bhhb = (const float*)d_in[75];
  const float *cw = (const float*)d_in[76], *cb = (const float*)d_in[77];
  float* out = (float*)d_out;

  // ---- workspace carve ----
  size_t off = 0;
  char* wsb = (char*)d_ws;
  auto carve = [&](size_t bytes) -> void* {
    void* p = wsb + off;
    off += (bytes + 255) & ~(size_t)255;
    return p;
  };
  const size_t SIGELTS = 38464;  // max per-sig activation elems (601*64)
  h16* XH  = (h16*)carve(5760000ull * 2);
  h16* BA  = (h16*)carve(1920ull * SIGELTS * 2);
  h16* BB  = (h16*)carve(1920ull * SIGELTS * 2);
  h16* BC  = (h16*)carve(1920ull * SIGELTS * 2);
  h16* PW1 = (h16*)carve(204800ull * 2);   // l1.conv2
  h16* PW2 = (h16*)carve(40960ull * 2);    // l2.conv1
  h16* PW3 = (h16*)carve(81920ull * 2);    // l2.conv2
  h16* PW4 = (h16*)carve(8192ull * 2);     // l2.shortcut
  h16* PW5 = (h16*)carve(98304ull * 2);    // l3.conv1
  h16* PW6 = (h16*)carve(196608ull * 2);   // l3.conv2
  h16* PW7 = (h16*)carve(32768ull * 2);    // l3.shortcut
  float* FEAT  = (float*)carve(1920ull * 128 * 4);
  float* LOGAP = (float*)carve(36 * 4);
  float* TW    = (float*)carve(36 * 4);
  float* FUSED = (float*)carve(320ull * 256 * 4);
  (void)ws_size; (void)in_sizes; (void)out_size;

  // ---- preprocessing ----
  cast_x_kernel<<<22500, 256, 0, stream>>>(x, XH, 5760000L);
  pack_w_kernel<<<(204800 + 255) / 256, 256, 0, stream>>>(l1w2, l1g2, PW1, 64, 64, 50);
  pack_w_kernel<<<(40960 + 255) / 256, 256, 0, stream>>>(l2w1, l2g1, PW2, 128, 64, 5);
  pack_w_kernel<<<(81920 + 255) / 256, 256, 0, stream>>>(l2w2, l2g2, PW3, 128, 128, 5);
  pack_w_kernel<<<(8192 + 255) / 256, 256, 0, stream>>>(l2ws, l2gs, PW4, 128, 64, 1);
  pack_w_kernel<<<(98304 + 255) / 256, 256, 0, stream>>>(l3w1, l3g1, PW5, 256, 128, 3);
  pack_w_kernel<<<(196608 + 255) / 256, 256, 0, stream>>>(l3w2, l3g2, PW6, 256, 256, 3);
  pack_w_kernel<<<(32768 + 255) / 256, 256, 0, stream>>>(l3ws, l3gs, PW7, 256, 128, 1);

  // ---- TCN stack ----
  l1_conv1_kernel<<<dim3(1920, 151), 256, 0, stream>>>(x, l1w1, l1b1, l1g1, l1be1, BA);

  ConvArgs a1{};   // layer1 tail: conv2(y1)+BN + scalar shortcut(x)+BN, ReLU
  a1.in = BA; a1.out = BB; a1.wpk = PW1; a1.bc = l1b2; a1.gc = l1g2; a1.bec = l1be2;
  a1.Cin = 64; a1.Lin = 601; a1.Cout = 64; a1.Lout = 600;
  a1.k = 50; a1.stride = 1; a1.dil = 1; a1.pad = 25;
  a1.scMode = 1; a1.scIn = XH; a1.scWraw = l1ws;
  a1.bs2 = l1bs; a1.gs2 = l1gs; a1.bes2 = l1bes;
  a1.Csc = 1; a1.Lsc = 3000; a1.scStride = 5;
  wmma_conv_kernel<<<dim3(1920, 4 * 38), 32, 0, stream>>>(a1);

  ConvArgs a2{};   // layer2 conv1 (mid)
  a2.in = BB; a2.out = BC; a2.wpk = PW2; a2.bc = l2b1; a2.gc = l2g1; a2.bec = l2be1;
  a2.Cin = 64; a2.Lin = 600; a2.Cout = 128; a2.Lout = 300;
  a2.k = 5; a2.stride = 2; a2.dil = 2; a2.pad = 4; a2.scMode = 0;
  wmma_conv_kernel<<<dim3(1920, 8 * 19), 32, 0, stream>>>(a2);

  ConvArgs a3{};   // layer2 tail
  a3.in = BC; a3.out = BA; a3.wpk = PW3; a3.bc = l2b2; a3.gc = l2g2; a3.bec = l2be2;
  a3.Cin = 128; a3.Lin = 300; a3.Cout = 128; a3.Lout = 300;
  a3.k = 5; a3.stride = 1; a3.dil = 1; a3.pad = 2;
  a3.scMode = 2; a3.scIn = BB; a3.scWpk = PW4;
  a3.bs2 = l2bs; a3.gs2 = l2gs; a3.bes2 = l2bes;
  a3.Csc = 64; a3.Lsc = 600; a3.scStride = 2;
  wmma_conv_kernel<<<dim3(1920, 8 * 19), 32, 0, stream>>>(a3);

  ConvArgs a4{};   // layer3 conv1 (mid)
  a4.in = BA; a4.out = BB; a4.wpk = PW5; a4.bc = l3b1; a4.gc = l3g1; a4.bec = l3be1;
  a4.Cin = 128; a4.Lin = 300; a4.Cout = 256; a4.Lout = 150;
  a4.k = 3; a4.stride = 2; a4.dil = 4; a4.pad = 4; a4.scMode = 0;
  wmma_conv_kernel<<<dim3(1920, 16 * 10), 32, 0, stream>>>(a4);

  ConvArgs a5{};   // layer3 tail
  a5.in = BB; a5.out = BC; a5.wpk = PW6; a5.bc = l3b2; a5.gc = l3g2; a5.bec = l3be2;
  a5.Cin = 256; a5.Lin = 150; a5.Cout = 256; a5.Lout = 150;
  a5.k = 3; a5.stride = 1; a5.dil = 1; a5.pad = 1;
  a5.scMode = 2; a5.scIn = BA; a5.scWpk = PW7;
  a5.bs2 = l3bs; a5.gs2 = l3gs; a5.bes2 = l3bes;
  a5.Csc = 128; a5.Lsc = 300; a5.scStride = 2;
  wmma_conv_kernel<<<dim3(1920, 16 * 10), 32, 0, stream>>>(a5);

  // ---- heads ----
  pool_fc_kernel<<<1920, 256, 0, stream>>>(BC, fcw, fcb, FEAT);
  setup_kernel<<<1, 64, 0, stream>>>(a_plv, pwf, pE, tpos, t1w, t1b, t2w, t2b, tatt,
                                     LOGAP, TW);
  gat_fuse_kernel<<<320, 256, 0, stream>>>(FEAT, twp, tbp, tatt, swp, sbp, satt, lam,
                                           sm1w, sm1b, sm2w, sm2b, mc1w, mc1b, mc2w, mc2b,
                                           ntg, ntb, nsg, nsb, gw, gb, LOGAP, TW,
                                           FUSED, out + 80);
  lstm_cls_kernel<<<16, 128, 0, stream>>>(FUSED, wihf, whhf, bihf, bhhf,
                                          wihb, whhb, bihb, bhhb, cw, cb, out);
}
